// GNN_Model_7103875908138
// MI455X (gfx1250) — matrix operand order repaired and checked by reading, not compiled, not run on hardware
//
#include <hip/hip_runtime.h>
#include <hip/hip_bf16.h>

// RouteNet-style GNN for MI455X (gfx1250, wave32, WMMA).
// States/weights in f16, accumulation in f32 via V_WMMA_F32_16X16X32_F16
// (K=32 == D, exact fit). One wave owns a 16-row tile of paths/links.

#define D32 32
#define TITER 8

typedef __attribute__((ext_vector_type(16))) _Float16 v16h;
typedef __attribute__((ext_vector_type(8)))  _Float16 v8h;
typedef __attribute__((ext_vector_type(8)))  float    v8f;

__device__ __forceinline__ v8f wmma16(v16h a, v16h b, v8f c) {
  // D = A(16x32 f16) * B(32x16 f16) + C(16x16 f32)
  return __builtin_amdgcn_wmma_f32_16x16x32_f16(false, a, false, b, (short)0, c,
                                                false, false);
}

__device__ __forceinline__ v8f zero8() {
  v8f v;
#pragma unroll
  for (int i = 0; i < 8; ++i) v[i] = 0.0f;
  return v;  // folds into inline-constant 0 as WMMA SRC2
}

// A-fragment (16x32 f16) loader. Lane l holds row M=l%16; half-select s=l/16
// picks K range [16s,16s+16). `row` must be 16B aligned.
__device__ __forceinline__ v16h load_a_frag(const _Float16* row, int s) {
  const v8h lo = *(const v8h*)(row + 16 * s);
  const v8h hi = *(const v8h*)(row + 16 * s + 8);
  v16h a;
#pragma unroll
  for (int i = 0; i < 8; ++i) { a[i] = lo[i]; a[i + 8] = hi[i]; }
  return a;
}

__device__ __forceinline__ v16h zero_a() {
  v16h a;
#pragma unroll
  for (int i = 0; i < 16; ++i) a[i] = (_Float16)0.0f;
  return a;
}

__device__ __forceinline__ float sigf(float x) { return 1.0f / (1.0f + __expf(-x)); }
__device__ __forceinline__ float tanhf_(float x) {
  float e = __expf(-2.0f * x);
  return (1.0f - e) / (1.0f + e);
}

// ---------------------------------------------------------------- weight pack
// B-fragment layout (32xK16 tile, wave32): lane l<16 -> N=l, halves e=K 0..15;
// lane l>=16 -> N=l-16, halves e=K 16..31. One tile = 32 lanes x 16 halves.
__device__ void packB_dev(const float* W, int N, int kbase, int nbase,
                          _Float16* dst, int tid, int nth) {
  for (int i = tid; i < 512; i += nth) {
    int lane = i >> 4, e = i & 15;
    int k = kbase + ((lane < 16) ? e : (16 + e));
    int n = nbase + (lane & 15);
    dst[i] = (_Float16)W[k * N + n];
  }
}

__global__ void pack_weights_kernel(const float* pathK, const float* pathR,
                                    const float* linkK, const float* linkR,
                                    const float* w1, const float* w2,
                                    _Float16* pPK, _Float16* pPR,
                                    _Float16* pLK, _Float16* pLR,
                                    _Float16* pW1, _Float16* pW2) {
  int tid = threadIdx.x, nth = blockDim.x;
  for (int t = 0; t < 6; ++t) {          // GRU kernels: [32 x 96], 6 N-tiles
    packB_dev(pathK, 96, 0, 16 * t, pPK + t * 512, tid, nth);
    packB_dev(pathR, 96, 0, 16 * t, pPR + t * 512, tid, nth);
    packB_dev(linkK, 96, 0, 16 * t, pLK + t * 512, tid, nth);
    packB_dev(linkR, 96, 0, 16 * t, pLR + t * 512, tid, nth);
  }
  for (int t = 0; t < 16; ++t)           // w1: [32 x 256]
    packB_dev(w1, 256, 0, 16 * t, pW1 + t * 512, tid, nth);
  for (int nt = 0; nt < 16; ++nt)        // w2: [256 x 256] -> 16 N x 8 K tiles
    for (int kc = 0; kc < 8; ++kc)
      packB_dev(w2, 256, 32 * kc, 16 * nt, pW2 + (size_t)(nt * 8 + kc) * 512,
                tid, nth);
}

// ------------------------------------------------------------------- prep
__global__ void init_states_kernel(const float* capacity, const float* traffic,
                                   const float* packets, const float* tdp,
                                   _Float16* ls16, _Float16* ps16, int* off,
                                   int* len, int n_links, int n_paths) {
  long i = (long)blockIdx.x * blockDim.x + threadIdx.x;
  if (i < (long)n_links * 32) {
    int l = (int)(i >> 5), d = (int)(i & 31);
    ls16[i] = (_Float16)(d == 0 ? capacity[l] : 0.0f);
  }
  if (i < (long)n_paths * 32) {
    int p = (int)(i >> 5), d = (int)(i & 31);
    float v = 0.0f;
    if (d == 0) v = traffic[p];
    else if (d == 1) v = packets[p];
    else if (d < 14) v = tdp[p * 12 + (d - 2)];
    ps16[i] = (_Float16)v;
  }
  if (i < n_paths) { off[i] = 0; len[i] = 0; }
}

__global__ void edges_meta_kernel(const int* path_ids, const int* seq, int* off,
                                  int* len, int E) {
  int e = blockIdx.x * blockDim.x + threadIdx.x;
  if (e >= E) return;
  int p = path_ids[e];
  if (seq[e] == 0) off[p] = e;
  if (e == E - 1 || path_ids[e + 1] != p) len[p] = seq[e] + 1;
}

__global__ void zero_accum_kernel(float* accum, int n) {
  int i = blockIdx.x * blockDim.x + threadIdx.x;
  if (i < n) accum[i] = 0.0f;
}

// segment_sum(path_state[path_of_edge], link_of_edge) -> accum[n_links][32]
__global__ void scatter_kernel(const _Float16* __restrict__ ps16,
                               const int* __restrict__ path_of_edge,
                               const int* __restrict__ link_of_edge,
                               float* __restrict__ accum, int E) {
  long t = (long)blockIdx.x * blockDim.x + threadIdx.x;
  int e = (int)(t >> 5), d = (int)(t & 31);
  if (e < E) {
    int p = path_of_edge[e];
    int l = link_of_edge[e];
    atomicAdd(&accum[(size_t)l * D32 + d], (float)ps16[(size_t)p * D32 + d]);
  }
}

// ------------------------------------------------------------ path GRU (RNN)
__global__ void __launch_bounds__(256) path_gru_kernel(
    _Float16* __restrict__ ps16, const _Float16* __restrict__ ls16,
    const _Float16* __restrict__ pWK, const _Float16* __restrict__ pWR,
    const float* __restrict__ bias, const int* __restrict__ link_to_path,
    const int* __restrict__ off, const int* __restrict__ len, int n_paths) {
  __shared__ __align__(16) _Float16 hbuf[8][16][40];  // per-wave h, padded rows
  const int lane = threadIdx.x & 31;
  const int wave = threadIdx.x >> 5;
  const int l16 = lane & 15;
  const int s = lane >> 4;
  const int Mb = s * 8;  // C-layout rows this lane holds

  // Register-resident weight B-fragments (reused every timestep / tile).
  v16h WK[6], WR[6];
#pragma unroll
  for (int t = 0; t < 6; ++t) {
    WK[t] = *(const v16h*)(pWK + t * 512 + lane * 16);
    WR[t] = *(const v16h*)(pWR + t * 512 + lane * 16);
  }
  float bz[2], br[2], bh_in[2], bh_rec[2];
#pragma unroll
  for (int j = 0; j < 2; ++j) {
    int n = 16 * j + l16;
    bz[j] = bias[n] + bias[96 + n];
    br[j] = bias[32 + n] + bias[96 + 32 + n];
    bh_in[j] = bias[64 + n];
    bh_rec[j] = bias[96 + 64 + n];
  }

  const int nTiles = (n_paths + 15) >> 4;
  const int wid = blockIdx.x * (blockDim.x >> 5) + wave;
  const int nw = gridDim.x * (blockDim.x >> 5);
  for (int tile = wid; tile < nTiles; tile += nw) {
    const int p0 = tile << 4;
    const int prow = p0 + l16;
    const bool rv = prow < n_paths;
    const int myOff = rv ? off[prow] : 0;
    const int myLen = rv ? len[prow] : 0;

    {  // stage h tile into LDS (row-major f16)
      const _Float16* src = ps16 + (size_t)prow * D32;
#pragma unroll
      for (int i = 0; i < 16; ++i)
        hbuf[wave][l16][16 * s + i] = rv ? src[16 * s + i] : (_Float16)0.0f;
    }
    int lenM[8];
#pragma unroll
    for (int m = 0; m < 8; ++m) {
      int pr = p0 + Mb + m;
      lenM[m] = (pr < n_paths) ? len[pr] : 0;
    }
    // h in C layout, register-resident across timesteps (same-wave LDS is
    // in-order so these reads see the staging stores above).
    float hC[2][8];
#pragma unroll
    for (int j = 0; j < 2; ++j)
#pragma unroll
      for (int m = 0; m < 8; ++m)
        hC[j][m] = (float)hbuf[wave][Mb + m][16 * j + l16];

    int tl = myLen;  // max length in tile (uniform after reduce)
#pragma unroll
    for (int o = 1; o < 32; o <<= 1) {
      int v = __shfl_xor(tl, o, 32);
      tl = tl > v ? tl : v;
    }

    for (int t = 0; t < tl; ++t) {
      // Branchless gather: clamp the step index; masked rows load a stale row
      // whose GRU result is discarded by the select below (Keras Masking).
      int tc = (t < myLen) ? t : (myLen - 1);
      tc = tc < 0 ? 0 : tc;
      const int lidx = link_to_path[myOff + tc];
      const v16h xa = load_a_frag(ls16 + (size_t)lidx * D32, s);
      {  // prefetch next timestep's link row into the near (WGP) cache
        int tn = (t + 1 < myLen) ? (t + 1) : tc;
        int lnext = link_to_path[myOff + tn];
        __builtin_prefetch(ls16 + (size_t)lnext * D32, 0, 3);
      }
      const v16h ha = load_a_frag(&hbuf[wave][l16][0], s);

      v8f az[2], ar[2], axh[2], ahh[2];
#pragma unroll
      for (int j = 0; j < 2; ++j) {  // 12 WMMAs; chains start from inline 0
        az[j]  = wmma16(ha, WR[j],     wmma16(xa, WK[j],     zero8()));
        ar[j]  = wmma16(ha, WR[2 + j], wmma16(xa, WK[2 + j], zero8()));
        axh[j] = wmma16(xa, WK[4 + j], zero8());
        ahh[j] = wmma16(ha, WR[4 + j], zero8());
      }
#pragma unroll
      for (int j = 0; j < 2; ++j) {
        const int col = 16 * j + l16;
#pragma unroll
        for (int m = 0; m < 8; ++m) {
          const float hold = hC[j][m];
          float z = sigf(az[j][m] + bz[j]);
          float r = sigf(ar[j][m] + br[j]);
          float c = tanhf_(axh[j][m] + bh_in[j] + r * (ahh[j][m] + bh_rec[j]));
          float hn = z * hold + (1.0f - z) * c;
          hn = (t < lenM[m]) ? hn : hold;  // v_cndmask, no exec branching
          hC[j][m] = hn;
          hbuf[wave][Mb + m][col] = (_Float16)hn;  // unconditional ds_store
        }
      }
      // same-wave LDS ops are in-order: no barrier needed before next step
    }
    if (rv) {
      _Float16* dst = ps16 + (size_t)prow * D32;
#pragma unroll
      for (int i = 0; i < 16; ++i) dst[16 * s + i] = hbuf[wave][l16][16 * s + i];
    }
  }
}

// ---------------------------------------------------------------- link GRU
template <bool FULL>
__device__ __forceinline__ void link_gru_tile(
    _Float16* __restrict__ ls16, const float* __restrict__ accum,
    const v16h* WK, const v16h* WR, const float* bz, const float* br,
    const float* bh_in, const float* bh_rec, int p0, int l16, int s, int Mb,
    int n_links) {
  const int row = p0 + l16;
  const bool rv = FULL || row < n_links;
  v16h xa, ha;
  if (rv) {
    const float* xr = accum + (size_t)row * D32;
#pragma unroll
    for (int i = 0; i < 8; ++i) {
      xa[i] = (_Float16)xr[16 * s + i];
      xa[8 + i] = (_Float16)xr[16 * s + 8 + i];
    }
    ha = load_a_frag(ls16 + (size_t)row * D32, s);
  } else {
    xa = zero_a();
    ha = zero_a();
  }
  v8f az[2], ar[2], axh[2], ahh[2];
#pragma unroll
  for (int j = 0; j < 2; ++j) {
    az[j]  = wmma16(ha, WR[j],     wmma16(xa, WK[j],     zero8()));
    ar[j]  = wmma16(ha, WR[2 + j], wmma16(xa, WK[2 + j], zero8()));
    axh[j] = wmma16(xa, WK[4 + j], zero8());
    ahh[j] = wmma16(ha, WR[4 + j], zero8());
  }
#pragma unroll
  for (int j = 0; j < 2; ++j) {
    const int col = 16 * j + l16;
#pragma unroll
    for (int m = 0; m < 8; ++m) {
      const int lr = p0 + Mb + m;
      if (FULL || lr < n_links) {
        float hold = (float)ls16[(size_t)lr * D32 + col];
        float z = sigf(az[j][m] + bz[j]);
        float r = sigf(ar[j][m] + br[j]);
        float c = tanhf_(axh[j][m] + bh_in[j] + r * (ahh[j][m] + bh_rec[j]));
        ls16[(size_t)lr * D32 + col] = (_Float16)(z * hold + (1.0f - z) * c);
      }
    }
  }
}

__global__ void __launch_bounds__(256) link_gru_kernel(
    _Float16* __restrict__ ls16, const float* __restrict__ accum,
    const _Float16* __restrict__ pWK, const _Float16* __restrict__ pWR,
    const float* __restrict__ bias, int n_links) {
  const int lane = threadIdx.x & 31;
  const int wave = threadIdx.x >> 5;
  const int l16 = lane & 15;
  const int s = lane >> 4;
  const int Mb = s * 8;

  v16h WK[6], WR[6];
#pragma unroll
  for (int t = 0; t < 6; ++t) {
    WK[t] = *(const v16h*)(pWK + t * 512 + lane * 16);
    WR[t] = *(const v16h*)(pWR + t * 512 + lane * 16);
  }
  float bz[2], br[2], bh_in[2], bh_rec[2];
#pragma unroll
  for (int j = 0; j < 2; ++j) {
    int n = 16 * j + l16;
    bz[j] = bias[n] + bias[96 + n];
    br[j] = bias[32 + n] + bias[96 + 32 + n];
    bh_in[j] = bias[64 + n];
    bh_rec[j] = bias[96 + 64 + n];
  }

  const int nTiles = (n_links + 15) >> 4;
  const int wid = blockIdx.x * (blockDim.x >> 5) + wave;
  const int nw = gridDim.x * (blockDim.x >> 5);
  for (int tile = wid; tile < nTiles; tile += nw) {
    const int p0 = tile << 4;
    if (p0 + 16 <= n_links)  // uniform branch; common case is guard-free
      link_gru_tile<true>(ls16, accum, WK, WR, bz, br, bh_in, bh_rec, p0, l16,
                          s, Mb, n_links);
    else
      link_gru_tile<false>(ls16, accum, WK, WR, bz, br, bh_in, bh_rec, p0, l16,
                           s, Mb, n_links);
  }
}

// ------------------------------------------------------------- readout MLP
__global__ void __launch_bounds__(256) readout_kernel(
    const _Float16* __restrict__ ps16, const _Float16* __restrict__ pW1,
    const float* __restrict__ b1, const _Float16* __restrict__ pW2,
    const float* __restrict__ b2, const float* __restrict__ w3,
    const float* __restrict__ b3, float* __restrict__ out, int n_paths) {
  __shared__ __align__(16) _Float16 h1buf[8][16][264];  // 16x256 + pad per wave
  const int lane = threadIdx.x & 31;
  const int wave = threadIdx.x >> 5;
  const int l16 = lane & 15;
  const int s = lane >> 4;
  const int Mb = s * 8;
  const float b3v = b3[0];

  const int nTiles = (n_paths + 15) >> 4;
  const int wid = blockIdx.x * (blockDim.x >> 5) + wave;
  const int nw = gridDim.x * (blockDim.x >> 5);
  for (int tile = wid; tile < nTiles; tile += nw) {
    const int p0 = tile << 4;
    const int prow = p0 + l16;
    const bool rv = prow < n_paths;
    v16h a0 = rv ? load_a_frag(ps16 + (size_t)prow * D32, s) : zero_a();

    // layer 1: h1 = relu(ps @ w1 + b1), 16 WMMAs -> LDS f16
    for (int nt = 0; nt < 16; ++nt) {
      v16h bf = *(const v16h*)(pW1 + (size_t)nt * 512 + lane * 16);
      v8f c = wmma16(a0, bf, zero8());
      const int col = 16 * nt + l16;
      const float b1n = b1[col];
#pragma unroll
      for (int m = 0; m < 8; ++m)
        h1buf[wave][Mb + m][col] = (_Float16)fmaxf(c[m] + b1n, 0.0f);
    }
    // preload h1 A-fragments (same-wave LDS in-order; no barrier)
    v16h a1[8];
#pragma unroll
    for (int kc = 0; kc < 8; ++kc)
      a1[kc] = load_a_frag(&h1buf[wave][l16][32 * kc], s);

    // layer 2 (128 WMMAs) fused with layer 3 dot product
    float pm[8];
#pragma unroll
    for (int m = 0; m < 8; ++m) pm[m] = 0.0f;
    for (int nt = 0; nt < 16; ++nt) {
      v8f c = zero8();
#pragma unroll
      for (int kc = 0; kc < 8; ++kc) {
        v16h bf = *(const v16h*)(pW2 + (size_t)(nt * 8 + kc) * 512 + lane * 16);
        c = wmma16(a1[kc], bf, c);
      }
      const float b2n = b2[16 * nt + l16];
      const float w3n = w3[16 * nt + l16];
#pragma unroll
      for (int m = 0; m < 8; ++m) pm[m] += fmaxf(c[m] + b2n, 0.0f) * w3n;
    }
#pragma unroll
    for (int o = 1; o < 16; o <<= 1)
#pragma unroll
      for (int m = 0; m < 8; ++m) pm[m] += __shfl_xor(pm[m], o, 32);
    if (l16 == 0) {
#pragma unroll
      for (int m = 0; m < 8; ++m) {
        int pr = p0 + Mb + m;
        if (pr < n_paths) out[pr] = pm[m] + b3v;
      }
    }
  }
}

// ---------------------------------------------------------------- launcher
extern "C" void kernel_launch(void* const* d_in, const int* in_sizes, int n_in,
                              void* d_out, int out_size, void* d_ws,
                              size_t ws_size, hipStream_t stream) {
  (void)n_in; (void)out_size; (void)ws_size;
  const float* capacity = (const float*)d_in[0];
  const float* traffic  = (const float*)d_in[1];
  const float* packets  = (const float*)d_in[2];
  const float* tdp      = (const float*)d_in[3];
  const float* pathK    = (const float*)d_in[4];
  const float* pathR    = (const float*)d_in[5];
  const float* pathB    = (const float*)d_in[6];
  const float* linkK    = (const float*)d_in[7];
  const float* linkR    = (const float*)d_in[8];
  const float* linkB    = (const float*)d_in[9];
  const float* w1 = (const float*)d_in[10];
  const float* b1 = (const float*)d_in[11];
  const float* w2 = (const float*)d_in[12];
  const float* b2 = (const float*)d_in[13];
  const float* w3 = (const float*)d_in[14];
  const float* b3 = (const float*)d_in[15];
  const int* link_to_path = (const int*)d_in[16];
  const int* path_ids     = (const int*)d_in[17];
  const int* seq_path     = (const int*)d_in[18];
  const int* path_to_link = (const int*)d_in[19];
  const int* seq_links    = (const int*)d_in[20];
  const int n_links = in_sizes[0];   // capacity: (n_links, 1)
  const int n_paths = in_sizes[1];   // traffic:  (n_paths, 1)
  const int E = in_sizes[16];        // link_to_path: (E,)

  char* w = (char*)d_ws;
  auto carve = [&](size_t bytes) -> void* {
    void* p = (void*)w;
    w += (bytes + 255) & ~(size_t)255;
    return p;
  };
  _Float16* ps16 = (_Float16*)carve((size_t)n_paths * D32 * 2);
  _Float16* ls16 = (_Float16*)carve((size_t)n_links * D32 * 2);
  float* accum = (float*)carve((size_t)n_links * D32 * 4);
  int* off = (int*)carve((size_t)n_paths * 4);
  int* len = (int*)carve((size_t)n_paths * 4);
  _Float16* pPK = (_Float16*)carve(6 * 512 * 2);
  _Float16* pPR = (_Float16*)carve(6 * 512 * 2);
  _Float16* pLK = (_Float16*)carve(6 * 512 * 2);
  _Float16* pLR = (_Float16*)carve(6 * 512 * 2);
  _Float16* pW1 = (_Float16*)carve(16 * 512 * 2);
  _Float16* pW2 = (_Float16*)carve(128 * 512 * 2);

  pack_weights_kernel<<<1, 256, 0, stream>>>(pathK, pathR, linkK, linkR, w1, w2,
                                             pPK, pPR, pLK, pLR, pW1, pW2);
  {
    long total = (long)n_paths * D32;
    long t2 = (long)n_links * D32;
    if (t2 > total) total = t2;
    init_states_kernel<<<(int)((total + 255) / 256), 256, 0, stream>>>(
        capacity, traffic, packets, tdp, ls16, ps16, off, len, n_links, n_paths);
  }
  edges_meta_kernel<<<(E + 255) / 256, 256, 0, stream>>>(path_ids, seq_path,
                                                         off, len, E);

  const int tilesP = (n_paths + 15) / 16;
  const int tilesL = (n_links + 15) / 16;
  const int blocksP = (tilesP + 7) / 8;
  const int blocksL = (tilesL + 7) / 8;
  for (int it = 0; it < TITER; ++it) {
    path_gru_kernel<<<blocksP, 256, 0, stream>>>(
        ps16, ls16, pPK, pPR, pathB, link_to_path, off, len, n_paths);
    zero_accum_kernel<<<(n_links * D32 + 255) / 256, 256, 0, stream>>>(
        accum, n_links * D32);
    long st = (long)E * D32;
    scatter_kernel<<<(int)((st + 255) / 256), 256, 0, stream>>>(
        ps16, path_to_link, seq_links, accum, E);
    link_gru_kernel<<<blocksL, 256, 0, stream>>>(ls16, accum, pLK, pLR, linkB,
                                                 n_links);
  }
  readout_kernel<<<blocksP, 256, 0, stream>>>(ps16, pW1, b1, pW2, b2, w3, b3,
                                              (float*)d_out, n_paths);
}
